// GraphReadout_47141561040925
// MI455X (gfx1250) — compile-verified, matrix operand order checked by reading, not run on hardware
//
#include <hip/hip_runtime.h>
#include <hip/hip_bf16.h>
#include <math.h>
#include <stdint.h>

typedef __attribute__((ext_vector_type(16))) _Float16 v16h;
typedef __attribute__((ext_vector_type(8)))  _Float16 v8h;
typedef __attribute__((ext_vector_type(8)))  float    v8f;

#define Nn   200000
#define Dd   256
#define Bb   1024
#define Hh   128
#define C1   768
#define C2   512
#define OUTD 256

// ---------------- workspace byte offsets ----------------
constexpr size_t OFF_POOL0 = 0;
constexpr size_t OFF_POOL1 = OFF_POOL0 + (size_t)Bb * Dd * 4;
constexpr size_t OFF_POOL2 = OFF_POOL1 + (size_t)Bb * Dd * 4;
constexpr size_t OFF_ATTNP = OFF_POOL2 + (size_t)Bb * Dd * 4;
constexpr size_t OFF_CNTT  = OFF_ATTNP + (size_t)Bb * Dd * 4;   // Bb*3 floats
constexpr size_t OFF_MKEY  = OFF_CNTT  + (size_t)Bb * 3 * 4;    // Bb u32
constexpr size_t OFF_Z     = OFF_MKEY  + (size_t)Bb * 4;        // Bb f32
constexpr size_t OFF_S     = OFF_Z     + (size_t)Bb * 4;        // Nn f32
constexpr size_t OFF_E     = OFF_S     + (size_t)Nn * 4;        // Nn f32
constexpr size_t OFF_COMB  = OFF_E     + (size_t)Nn * 4;        // Bb*C1 f16
constexpr size_t OFF_AW1T  = OFF_COMB  + (size_t)Bb * C1 * 2;   // Hh*Dd f16  [n][k]
constexpr size_t OFF_WET   = OFF_AW1T  + (size_t)Hh * Dd * 2;   // Dd*Dd f16
constexpr size_t OFF_WVT   = OFF_WET   + (size_t)Dd * Dd * 2;
constexpr size_t OFF_WPT   = OFF_WVT   + (size_t)Dd * Dd * 2;
constexpr size_t OFF_W1T   = OFF_WPT   + (size_t)Dd * Dd * 2;   // C2*C1 f16
constexpr size_t OFF_W2T   = OFF_W1T   + (size_t)C2 * C1 * 2;   // OUTD*C2 f16
constexpr size_t ZERO_WORDS = OFF_S / 4;  // zero pools/cnts/mkey/z each call

// ---------------- CDNA5 feature probes ----------------
#if defined(__HIP_DEVICE_COMPILE__) && __has_builtin(__builtin_amdgcn_s_wait_asynccnt)
#define HAVE_ASYNC_LDS 1
#endif

#ifdef HAVE_ASYNC_LDS
__device__ __forceinline__ void async_cp16(const void* g, void* l) {
  // GLOBAL_LOAD_ASYNC_TO_LDS_B128: VDST = per-lane LDS byte address,
  // VADDR = per-lane 64-bit global address, saddr = off.
  // Generic LDS pointers carry the LDS byte offset in their low 32 bits.
  uint32_t loff = (uint32_t)(uintptr_t)l;
  uint64_t ga   = (uint64_t)(uintptr_t)g;
  asm volatile("global_load_async_to_lds_b128 %0, %1, off"
               :
               : "v"(loff), "v"(ga)
               : "memory");
}
__device__ __forceinline__ void async_wait0() {
  __builtin_amdgcn_s_wait_asynccnt(0);
}
#endif

#if defined(__HIP_DEVICE_COMPILE__) && __has_builtin(__builtin_amdgcn_tanhf)
#define fast_tanh(xx) __builtin_amdgcn_tanhf(xx)
#elif defined(__HIP_DEVICE_COMPILE__) && __has_builtin(__builtin_amdgcn_tanh_f32)
#define fast_tanh(xx) __builtin_amdgcn_tanh_f32(xx)
#else
#define fast_tanh(xx) tanhf(xx)
#endif

// Butterfly sum over each 16-lane half using DS_SWIZZLE XOR patterns
// (group-of-32 mode: offset = xor_mask<<10 | and_mask 0x1f) — avoids the
// guard ALU that generic __shfl_xor generates around ds_bpermute.
__device__ __forceinline__ float xor_add16(float t) {
#if defined(__HIP_DEVICE_COMPILE__) && __has_builtin(__builtin_amdgcn_ds_swizzle)
  t += __int_as_float(__builtin_amdgcn_ds_swizzle(__float_as_int(t), 0x041F)); // ^1
  t += __int_as_float(__builtin_amdgcn_ds_swizzle(__float_as_int(t), 0x081F)); // ^2
  t += __int_as_float(__builtin_amdgcn_ds_swizzle(__float_as_int(t), 0x101F)); // ^4
  t += __int_as_float(__builtin_amdgcn_ds_swizzle(__float_as_int(t), 0x201F)); // ^8
#else
  t += __shfl_xor(t, 1, 32);
  t += __shfl_xor(t, 2, 32);
  t += __shfl_xor(t, 4, 32);
  t += __shfl_xor(t, 8, 32);
#endif
  return t;
}

// ---------------- helpers ----------------
__device__ __forceinline__ v16h frag_ld(const _Float16* p) {
  // CDNA5 16-bit fragment: per lane, halfs [0..7] = K base..base+7,
  // halfs [8..15] = K base+16..base+23 (base = 0 for lanes 0-15, 8 for 16-31).
  v8h lo = *(const v8h*)(p);
  v8h hv = *(const v8h*)(p + 16);
  return __builtin_shufflevector(lo, hv, 0,1,2,3,4,5,6,7,8,9,10,11,12,13,14,15);
}

__device__ __forceinline__ v8f wmma16(v16h a, v16h b, v8f c) {
  return __builtin_amdgcn_wmma_f32_16x16x32_f16(false, a, false, b, (short)0, c, false, false);
}

__device__ __forceinline__ unsigned fkey(float f) {
  unsigned u = __float_as_uint(f);
  return (u & 0x80000000u) ? ~u : (u | 0x80000000u);
}
__device__ __forceinline__ float keyf(unsigned k) {
  unsigned u = (k & 0x80000000u) ? (k & 0x7fffffffu) : ~k;
  return __uint_as_float(u);
}

// ---------------- K0: zero accumulators + f16 transposed weights ----------------
__global__ void k0_prep(const float* __restrict__ aw1, const float* __restrict__ we,
                        const float* __restrict__ wvv, const float* __restrict__ wpp,
                        const float* __restrict__ w1,  const float* __restrict__ w2,
                        unsigned* __restrict__ zbase,
                        _Float16* __restrict__ aw1T, _Float16* __restrict__ weT,
                        _Float16* __restrict__ wvT,  _Float16* __restrict__ wpT,
                        _Float16* __restrict__ w1T,  _Float16* __restrict__ w2T) {
  size_t gid = (size_t)blockIdx.x * blockDim.x + threadIdx.x;
  size_t gsz = (size_t)gridDim.x * blockDim.x;
  for (size_t i = gid; i < ZERO_WORDS; i += gsz) zbase[i] = 0u;
  for (size_t i = gid; i < (size_t)Hh * Dd; i += gsz) {
    size_t n = i / Dd, k = i % Dd;
    aw1T[i] = (_Float16)aw1[k * Hh + n];
  }
  for (size_t i = gid; i < (size_t)Dd * Dd; i += gsz) {
    size_t n = i / Dd, k = i % Dd;
    weT[i] = (_Float16)we[k * Dd + n];
    wvT[i] = (_Float16)wvv[k * Dd + n];
    wpT[i] = (_Float16)wpp[k * Dd + n];
  }
  for (size_t i = gid; i < (size_t)C2 * C1; i += gsz) {
    size_t n = i / C1, k = i % C1;
    w1T[i] = (_Float16)w1[k * C2 + n];
  }
  for (size_t i = gid; i < (size_t)OUTD * C2; i += gsz) {
    size_t n = i / C2, k = i % C2;
    w2T[i] = (_Float16)w2[k * OUTD + n];
  }
}

// ---------------- K1: fused pass 1 ----------------
// 256 contiguous nodes per block. x streamed in double-buffered 16x256 LDS
// tiles via async global->LDS DMA (overlapped with compute). Thread t owns
// column t for segmented typed-pool accumulation (atomic flush on segment
// boundary); each 16-row tile is converted to f16 and hit with WMMA against
// resident aw1^T fragments to produce attention scores.
__global__ __launch_bounds__(256) void k1_pass1(
    const float* __restrict__ x, const int* __restrict__ batch,
    const int* __restrict__ mask,
    const float* __restrict__ ab1, const float* __restrict__ aw2,
    const float* __restrict__ ab2, const _Float16* __restrict__ aw1T,
    float* __restrict__ pool0, float* __restrict__ pool1, float* __restrict__ pool2,
    float* __restrict__ cntt, float* __restrict__ sOut) {
  __shared__ __attribute__((aligned(16))) float    Xs[2][16][Dd];
  __shared__ __attribute__((aligned(16))) _Float16 Atile[16][264];
  __shared__ float s_red[16][8];
  __shared__ int   sh_seg[16];
  __shared__ int   sh_ty[16];

  const int tid  = threadIdx.x;
  const int lane = tid & 31;
  const int wvi  = tid >> 5;
  const int lrow = lane & 15;
  const int hsel = (lane >> 4) & 1;
  const int r0   = blockIdx.x * 256;

  // Resident B fragments: wave wvi covers score cols [16*wvi, 16*wvi+16)
  v16h Bf[8];
  {
    const _Float16* bp = aw1T + (size_t)(16 * wvi + lrow) * Dd;
#pragma unroll
    for (int ks = 0; ks < 8; ++ks) Bf[ks] = frag_ld(bp + 32 * ks + hsel * 8);
  }
  const int   ncol = 16 * wvi + lrow;  // 0..127
  const float a2   = aw2[ncol];
  const float b1v  = ab1[ncol];
  const float b2v  = ab2[0];

  const int rows_left = Nn - r0;
  const int nch = (rows_left >= 256) ? 16 : (rows_left + 15) / 16;

  auto stage = [&](int c, int buf) {
    const int r = r0 + 16 * c;
#ifdef HAVE_ASYNC_LDS
    if (r + 16 <= Nn) {
      const float* gsrc = x + (size_t)r * Dd;
      float*       ldst = &Xs[buf][0][0];
#pragma unroll
      for (int q = 0; q < 4; ++q)
        async_cp16(gsrc + (q * 256 + tid) * 4, ldst + (q * 256 + tid) * 4);
      return;
    }
#endif
#pragma unroll
    for (int k = 0; k < 16; ++k) {
      int rr = r + k;
      Xs[buf][k][tid] = (rr < Nn) ? x[(size_t)rr * Dd + tid] : 0.f;
    }
  };

  float acc0 = 0.f, acc1 = 0.f, acc2v = 0.f, cntf = 0.f;
  int   cur  = -1;

  stage(0, 0);
  for (int c = 0; c < nch; ++c) {
    const int r   = r0 + c * 16;
    const int buf = c & 1;
#ifdef HAVE_ASYNC_LDS
    async_wait0();
#endif
    __syncthreads();  // chunk c visible everywhere; prior-chunk LDS reads done
    if (c + 1 < nch) stage(c + 1, buf ^ 1);  // overlap next DMA with compute
    if (tid < 16) {
      int rr      = r + tid;
      sh_seg[tid] = (rr < Nn) ? batch[rr] : -1;
      int ty      = 0;
      if (rr < Nn) ty = mask[rr * 3 + 1] + 2 * mask[rr * 3 + 2];
      sh_ty[tid] = ty;
    }
    __syncthreads();  // sh_seg/sh_ty visible; Atile writable

    float xv[16];
#pragma unroll
    for (int k = 0; k < 16; ++k) xv[k] = Xs[buf][k][tid];

#pragma unroll
    for (int k = 0; k < 16; ++k) {
      int sg = sh_seg[k];
      if (sg >= 0) {
        if (sg != cur) {
          if (cur >= 0) {
            atomicAdd(pool0 + (size_t)cur * Dd + tid, acc0);
            atomicAdd(pool1 + (size_t)cur * Dd + tid, acc1);
            atomicAdd(pool2 + (size_t)cur * Dd + tid, acc2v);
            if (tid < 3) atomicAdd(cntt + cur * 3 + tid, cntf);
          }
          acc0 = acc1 = acc2v = 0.f;
          cntf = 0.f;
          cur  = sg;
        }
        int ty = sh_ty[k];
        if (ty == 0) acc0 += xv[k];
        else if (ty == 1) acc1 += xv[k];
        else acc2v += xv[k];
        if (tid < 3 && ty == tid) cntf += 1.0f;
      }
      Atile[k][tid] = (_Float16)xv[k];
    }
    __syncthreads();  // Atile ready for WMMA

    v8f  acc = {};
    v16h af  = frag_ld(&Atile[lrow][hsel * 8]);
#pragma unroll
    for (int ks = 0; ks < 8; ++ks) {
      v16h nxt = af;
      if (ks < 7) nxt = frag_ld(&Atile[lrow][32 * (ks + 1) + hsel * 8]);
      acc = wmma16(af, Bf[ks], acc);
      af  = nxt;
    }
#pragma unroll
    for (int v = 0; v < 8; ++v) {
      float t = fast_tanh(acc[v] + b1v) * a2;
      t       = xor_add16(t);
      if (lrow == 0) s_red[v + 8 * hsel][wvi] = t;
    }
    __syncthreads();
    if (tid < 16) {
      int rr = r + tid;
      if (rr < Nn) {
        float ssum = b2v;
#pragma unroll
        for (int q = 0; q < 8; ++q) ssum += s_red[tid][q];
        sOut[rr] = ssum;
      }
    }
  }
  if (cur >= 0) {
    atomicAdd(pool0 + (size_t)cur * Dd + tid, acc0);
    atomicAdd(pool1 + (size_t)cur * Dd + tid, acc1);
    atomicAdd(pool2 + (size_t)cur * Dd + tid, acc2v);
    if (tid < 3) atomicAdd(cntt + cur * 3 + tid, cntf);
  }
}

// ---------------- K2: segmented softmax normalization ----------------
__global__ void k2_max(const float* __restrict__ s, const int* __restrict__ batch,
                       unsigned* __restrict__ mkey) {
  int i = blockIdx.x * 256 + threadIdx.x;
  if (i < Nn) atomicMax(mkey + batch[i], fkey(s[i]));
}

__global__ void k2_exp(const float* __restrict__ s, const int* __restrict__ batch,
                       const unsigned* __restrict__ mkey, float* __restrict__ e,
                       float* __restrict__ z) {
  int i = blockIdx.x * 256 + threadIdx.x;
  if (i < Nn) {
    int   b  = batch[i];
    float ev = __expf(s[i] - keyf(mkey[b]));
    e[i]     = ev;
    atomicAdd(z + b, ev);
  }
}

// ---------------- Kp2: pass 2, attention-weighted segmented sum ----------------
__global__ __launch_bounds__(256) void k_attnpool(
    const float* __restrict__ x, const int* __restrict__ batch,
    const float* __restrict__ e, const float* __restrict__ z,
    float* __restrict__ attnp) {
  __shared__ __attribute__((aligned(16))) float Xs[2][16][Dd];
  __shared__ int   sh_seg[16];
  __shared__ float sh_w[16];
  const int tid = threadIdx.x;
  const int r0  = blockIdx.x * 256;

  const int rows_left = Nn - r0;
  const int nch = (rows_left >= 256) ? 16 : (rows_left + 15) / 16;

  auto stage = [&](int c, int buf) {
    const int r = r0 + 16 * c;
#ifdef HAVE_ASYNC_LDS
    if (r + 16 <= Nn) {
      const float* gsrc = x + (size_t)r * Dd;
      float*       ldst = &Xs[buf][0][0];
#pragma unroll
      for (int q = 0; q < 4; ++q)
        async_cp16(gsrc + (q * 256 + tid) * 4, ldst + (q * 256 + tid) * 4);
      return;
    }
#endif
#pragma unroll
    for (int k = 0; k < 16; ++k) {
      int rr = r + k;
      Xs[buf][k][tid] = (rr < Nn) ? x[(size_t)rr * Dd + tid] : 0.f;
    }
  };

  float acc = 0.f;
  int   cur = -1;

  stage(0, 0);
  for (int c = 0; c < nch; ++c) {
    const int r   = r0 + c * 16;
    const int buf = c & 1;
#ifdef HAVE_ASYNC_LDS
    async_wait0();
#endif
    __syncthreads();
    if (c + 1 < nch) stage(c + 1, buf ^ 1);
    if (tid < 16) {
      int rr = r + tid;
      if (rr < Nn) {
        int b       = batch[rr];
        sh_seg[tid] = b;
        sh_w[tid]   = e[rr] / z[b];
      } else {
        sh_seg[tid] = -1;
        sh_w[tid]   = 0.f;
      }
    }
    __syncthreads();
#pragma unroll
    for (int k = 0; k < 16; ++k) {
      int sg = sh_seg[k];
      if (sg >= 0) {
        if (sg != cur) {
          if (cur >= 0) atomicAdd(attnp + (size_t)cur * Dd + tid, acc);
          acc = 0.f;
          cur = sg;
        }
        acc += sh_w[k] * Xs[buf][k][tid];
      }
    }
  }
  if (cur >= 0) atomicAdd(attnp + (size_t)cur * Dd + tid, acc);
}

// ---------------- K3a: typed-pool GEMMs + assemble combined (f16) ----------------
__global__ __launch_bounds__(256) void k3a_phys(
    const float* __restrict__ pool0, const float* __restrict__ pool1,
    const float* __restrict__ pool2, const float* __restrict__ cntt,
    const float* __restrict__ attnp,
    const float* __restrict__ be, const float* __restrict__ bvv,
    const float* __restrict__ bpp,
    const _Float16* __restrict__ weT, const _Float16* __restrict__ wvT,
    const _Float16* __restrict__ wpT, _Float16* __restrict__ comb) {
  __shared__ __attribute__((aligned(16))) _Float16 At[3][16][264];
  __shared__ float cnt_sh[16][4];
  const int tid  = threadIdx.x;
  const int lane = tid & 31;
  const int wvi  = tid >> 5;
  const int lrow = lane & 15;
  const int hsel = (lane >> 4) & 1;
  const int s0   = blockIdx.x * 16;

  if (tid < 16) {
    float c0 = cntt[(s0 + tid) * 3 + 0];
    float c1 = cntt[(s0 + tid) * 3 + 1];
    float c2 = cntt[(s0 + tid) * 3 + 2];
    cnt_sh[tid][0] = c0; cnt_sh[tid][1] = c1; cnt_sh[tid][2] = c2;
    cnt_sh[tid][3] = c0 + c1 + c2;
  }
  __syncthreads();
  for (int j = 0; j < 16; ++j) {
    size_t ro = (size_t)(s0 + j) * Dd + tid;
    float  p0 = pool0[ro], p1 = pool1[ro], p2 = pool2[ro];
    At[0][j][tid] = (_Float16)p0;
    At[1][j][tid] = (_Float16)p1;
    At[2][j][tid] = (_Float16)p2;
    comb[(size_t)(s0 + j) * C1 + tid]      = (_Float16)((p0 + p1 + p2) / cnt_sh[j][3]);
    comb[(size_t)(s0 + j) * C1 + Dd + tid] = (_Float16)attnp[ro];
  }
  __syncthreads();

  const _Float16* WT[3] = {weT, wvT, wpT};
  v8f acc[2] = {};
#pragma unroll
  for (int t = 0; t < 3; ++t) {
#pragma unroll
    for (int ks = 0; ks < 8; ++ks) {
      v16h af = frag_ld(&At[t][lrow][32 * ks + hsel * 8]);
#pragma unroll
      for (int j = 0; j < 2; ++j) {
        int  nt = 2 * wvi + j;
        v16h bf = frag_ld(WT[t] + (size_t)(16 * nt + lrow) * Dd + 32 * ks + hsel * 8);
        acc[j]  = wmma16(af, bf, acc[j]);
      }
    }
  }
#pragma unroll
  for (int j = 0; j < 2; ++j) {
    int   n   = 16 * (2 * wvi + j) + lrow;
    float bb0 = be[n], bb1 = bvv[n], bb2 = bpp[n];
#pragma unroll
    for (int v = 0; v < 8; ++v) {
      int   rw  = v + 8 * hsel;
      float val = acc[j][v] + cnt_sh[rw][0] * bb0 + cnt_sh[rw][1] * bb1 + cnt_sh[rw][2] * bb2;
      comb[(size_t)(s0 + rw) * C1 + 2 * Dd + n] = (_Float16)val;
    }
  }
}

// ---------------- K3b: fused silu MLP (768 -> 512 -> 256) ----------------
__global__ __launch_bounds__(256) void k3b_mlp(
    const _Float16* __restrict__ comb, const _Float16* __restrict__ w1T,
    const _Float16* __restrict__ w2T, const float* __restrict__ b1,
    const float* __restrict__ b2, float* __restrict__ out) {
  __shared__ __attribute__((aligned(16))) _Float16 Hs[16][520];
  const int tid  = threadIdx.x;
  const int lane = tid & 31;
  const int wvi  = tid >> 5;
  const int lrow = lane & 15;
  const int hsel = (lane >> 4) & 1;
  const int s0   = blockIdx.x * 16;

  v8f acc[4] = {};
#pragma unroll
  for (int ks = 0; ks < 24; ++ks) {
    v16h af = frag_ld(comb + (size_t)(s0 + lrow) * C1 + 32 * ks + hsel * 8);
#pragma unroll
    for (int j = 0; j < 4; ++j) {
      int  nt = 4 * wvi + j;
      v16h bf = frag_ld(w1T + (size_t)(16 * nt + lrow) * C1 + 32 * ks + hsel * 8);
      acc[j]  = wmma16(af, bf, acc[j]);
    }
  }
#pragma unroll
  for (int j = 0; j < 4; ++j) {
    int   n  = 16 * (4 * wvi + j) + lrow;
    float bb = b1[n];
#pragma unroll
    for (int v = 0; v < 8; ++v) {
      int   rw  = v + 8 * hsel;
      float val = acc[j][v] + bb;
      float sg  = 1.0f / (1.0f + __expf(-val));
      Hs[rw][n] = (_Float16)(val * sg);
    }
  }
  __syncthreads();
  v8f acc2[2] = {};
#pragma unroll
  for (int ks = 0; ks < 16; ++ks) {
    v16h af = frag_ld(&Hs[lrow][32 * ks + hsel * 8]);
#pragma unroll
    for (int j = 0; j < 2; ++j) {
      int  nt = 2 * wvi + j;
      v16h bf = frag_ld(w2T + (size_t)(16 * nt + lrow) * C2 + 32 * ks + hsel * 8);
      acc2[j] = wmma16(af, bf, acc2[j]);
    }
  }
#pragma unroll
  for (int j = 0; j < 2; ++j) {
    int   n  = 16 * (2 * wvi + j) + lrow;
    float bb = b2[n];
#pragma unroll
    for (int v = 0; v < 8; ++v) {
      int rw = v + 8 * hsel;
      out[(size_t)(s0 + rw) * OUTD + n] = acc2[j][v] + bb;
    }
  }
}

// ---------------- host launcher ----------------
extern "C" void kernel_launch(void* const* d_in, const int* in_sizes, int n_in,
                              void* d_out, int out_size, void* d_ws, size_t ws_size,
                              hipStream_t stream) {
  (void)in_sizes; (void)n_in; (void)out_size; (void)ws_size;
  const float* x     = (const float*)d_in[0];
  const int*   batch = (const int*)d_in[1];
  const int*   ntm   = (const int*)d_in[2];
  const float* aw1   = (const float*)d_in[4];
  const float* ab1   = (const float*)d_in[5];
  const float* aw2   = (const float*)d_in[6];
  const float* ab2   = (const float*)d_in[7];
  const float* we    = (const float*)d_in[8];
  const float* be    = (const float*)d_in[9];
  const float* wv    = (const float*)d_in[10];
  const float* bv    = (const float*)d_in[11];
  const float* wp    = (const float*)d_in[12];
  const float* bp    = (const float*)d_in[13];
  const float* w1    = (const float*)d_in[14];
  const float* b1    = (const float*)d_in[15];
  const float* w2    = (const float*)d_in[16];
  const float* b2    = (const float*)d_in[17];

  char* ws = (char*)d_ws;
  float*     pool0 = (float*)(ws + OFF_POOL0);
  float*     pool1 = (float*)(ws + OFF_POOL1);
  float*     pool2 = (float*)(ws + OFF_POOL2);
  float*     attnp = (float*)(ws + OFF_ATTNP);
  float*     cntt  = (float*)(ws + OFF_CNTT);
  unsigned*  mkey  = (unsigned*)(ws + OFF_MKEY);
  float*     zseg  = (float*)(ws + OFF_Z);
  float*     sval  = (float*)(ws + OFF_S);
  float*     eval  = (float*)(ws + OFF_E);
  _Float16*  comb  = (_Float16*)(ws + OFF_COMB);
  _Float16*  aw1T  = (_Float16*)(ws + OFF_AW1T);
  _Float16*  weT   = (_Float16*)(ws + OFF_WET);
  _Float16*  wvT   = (_Float16*)(ws + OFF_WVT);
  _Float16*  wpT   = (_Float16*)(ws + OFF_WPT);
  _Float16*  w1T   = (_Float16*)(ws + OFF_W1T);
  _Float16*  w2T   = (_Float16*)(ws + OFF_W2T);

  const int nodeBlocks = (Nn + 255) / 256;

  k0_prep<<<2048, 256, 0, stream>>>(aw1, we, wv, wp, w1, w2, (unsigned*)ws,
                                    aw1T, weT, wvT, wpT, w1T, w2T);
  k1_pass1<<<nodeBlocks, 256, 0, stream>>>(x, batch, ntm, ab1, aw2, ab2, aw1T,
                                           pool0, pool1, pool2, cntt, sval);
  k2_max<<<nodeBlocks, 256, 0, stream>>>(sval, batch, mkey);
  k2_exp<<<nodeBlocks, 256, 0, stream>>>(sval, batch, mkey, eval, zseg);
  k_attnpool<<<nodeBlocks, 256, 0, stream>>>(x, batch, eval, zseg, attnp);
  k3a_phys<<<Bb / 16, 256, 0, stream>>>(pool0, pool1, pool2, cntt, attnp,
                                        be, bv, bp, weT, wvT, wpT, comb);
  k3b_mlp<<<Bb / 16, 256, 0, stream>>>(comb, w1T, w2T, b1, b2, (float*)d_out);
}
